// PointNetFeatureUpsampling_49478023250591
// MI455X (gfx1250) — compile-verified, matrix operand order checked by reading, not run on hardware
//
#include <hip/hip_runtime.h>
#include <hip/hip_bf16.h>

typedef __bf16 bf16_t;
typedef __attribute__((ext_vector_type(16))) __bf16 v16bf;
typedef __attribute__((ext_vector_type(8)))  float  v8f;

#define F32_EPS 1.1920928955078125e-07f
#define BN_EPS  1e-5f

static __device__ __forceinline__ bf16_t f2bf(float f) {
    unsigned u = __builtin_bit_cast(unsigned, f);
    unsigned r = u + 0x7FFFu + ((u >> 16) & 1u);   // round-to-nearest-even
    unsigned short h = (unsigned short)(r >> 16);
    return __builtin_bit_cast(bf16_t, h);
}

// CDNA5 async global->LDS copy (ASYNCcnt-tracked, 16B per lane).
// ldsOff = LDS byte offset (low 32 bits of shared-aperture flat address).
static __device__ __forceinline__ void async_ld_b128(unsigned ldsOff, const void* gaddr) {
    asm volatile("global_load_async_to_lds_b128 %0, %1, off"
                 :: "v"(ldsOff), "v"(gaddr) : "memory");
}
#define S_WAIT_ASYNCCNT(n) asm volatile("s_wait_asynccnt %0" :: "n"(n) : "memory")

// ---------------------------------------------------------------------------
// Kernel 1: convert W1 [256x384] and W2 [128x256] f32 -> bf16
// ---------------------------------------------------------------------------
__global__ void cvt_weights_kernel(const float* __restrict__ W1, const float* __restrict__ W2,
                                   bf16_t* __restrict__ W1b, bf16_t* __restrict__ W2b) {
    int i = blockIdx.x * blockDim.x + threadIdx.x;
    if (i < 256 * 384) W1b[i] = f2bf(W1[i]);
    if (i < 128 * 256) W2b[i] = f2bf(W2[i]);
}

// ---------------------------------------------------------------------------
// Kernel 2: 3-NN over S=2048 candidates; xyz2 batch staged in LDS (24 KB).
// ---------------------------------------------------------------------------
__global__ __launch_bounds__(256) void knn3_kernel(const float* __restrict__ xyz1,
                                                   const float* __restrict__ xyz2,
                                                   int* __restrict__ outIdx,
                                                   float* __restrict__ outW) {
    const int N = 8192, S = 2048;
    __shared__ float sx[2048], sy[2048], sz[2048];
    int b    = blockIdx.x / (N / 256);
    int nblk = blockIdx.x % (N / 256);
    const float* x2 = xyz2 + (size_t)b * S * 3;
    for (int i = threadIdx.x; i < S; i += 256) {
        sx[i] = x2[i * 3 + 0]; sy[i] = x2[i * 3 + 1]; sz[i] = x2[i * 3 + 2];
    }
    __syncthreads();

    int n = nblk * 256 + threadIdx.x;
    const float* p = xyz1 + ((size_t)b * N + n) * 3;
    float px = p[0], py = p[1], pz = p[2];
    float d0 = 3.4e38f, d1 = 3.4e38f, d2 = 3.4e38f;
    int   i0 = 0, i1 = 0, i2 = 0;
    for (int s = 0; s < S; ++s) {
        float dx = px - sx[s], dy = py - sy[s], dz = pz - sz[s];
        float d = dx * dx + dy * dy + dz * dz;
        if (d < d0)      { d2 = d1; i2 = i1; d1 = d0; i1 = i0; d0 = d; i0 = s; }
        else if (d < d1) { d2 = d1; i2 = i1; d1 = d;  i1 = s; }
        else if (d < d2) { d2 = d;  i2 = s; }
    }
    float r0 = 1.0f / (sqrtf(fmaxf(d0, 0.0f)) + F32_EPS);
    float r1 = 1.0f / (sqrtf(fmaxf(d1, 0.0f)) + F32_EPS);
    float r2 = 1.0f / (sqrtf(fmaxf(d2, 0.0f)) + F32_EPS);
    float inv = 1.0f / (r0 + r1 + r2);
    size_t pt = (size_t)b * N + n;
    outIdx[pt * 3 + 0] = i0; outIdx[pt * 3 + 1] = i1; outIdx[pt * 3 + 2] = i2;
    outW[pt * 3 + 0] = r0 * inv; outW[pt * 3 + 1] = r1 * inv; outW[pt * 3 + 2] = r2 * inv;
}

// ---------------------------------------------------------------------------
// Kernel 3: interpolate points2 at 3 neighbors + concat with points1 -> bf16
// ---------------------------------------------------------------------------
__global__ __launch_bounds__(256) void interp_concat_kernel(const float* __restrict__ points1,
                                                            const float* __restrict__ points2,
                                                            const int* __restrict__ idx,
                                                            const float* __restrict__ w,
                                                            bf16_t* __restrict__ X) {
    const int N = 8192, S = 2048, D1 = 128, D2 = 256, Cin = 384;
    size_t pt = blockIdx.x;
    int b = (int)(pt / N);
    int c = threadIdx.x;
    const int*   id = idx + pt * 3;
    const float* ww = w + pt * 3;
    const float* P2 = points2 + (size_t)b * S * D2;
    float v = ww[0] * P2[(size_t)id[0] * D2 + c]
            + ww[1] * P2[(size_t)id[1] * D2 + c]
            + ww[2] * P2[(size_t)id[2] * D2 + c];
    bf16_t* row = X + pt * Cin;
    row[D1 + c] = f2bf(v);
    if (c < D1) row[c] = f2bf(points1[pt * D1 + c]);
}

// ---------------------------------------------------------------------------
// Kernel 4: bf16 WMMA GEMM  Y[M,N] = A[M,K] * W[N,K]^T  (f32 accumulate)
// Block: 256 threads (8 waves), M-tile 128, full N per block.
// Double-buffered LDS K-chunks (KC=32) staged via GLOBAL_LOAD_ASYNC_TO_LDS_B128
// (ASYNCcnt); next chunk's copies are in flight during the current WMMA chain.
// B fragments are software-pipelined two deep so dscnt waits overlap WMMA issue.
// ---------------------------------------------------------------------------
union FragBF { uint4 q[2]; v16bf v; };

template <int K, int NT>
__global__ __launch_bounds__(256) void gemm_bf16_wmma_kernel(const bf16_t* __restrict__ A,
                                                             const bf16_t* __restrict__ W,
                                                             float* __restrict__ Y) {
    constexpr int N      = NT * 16;
    constexpr int KC     = 32;
    constexpr int NCHUNK = K / KC;
    constexpr int AI     = (128 * KC) / (256 * 8);   // async b128 issues/thread for A (=2)
    constexpr int BI     = (N * KC) / (256 * 8);     // async b128 issues/thread for B (4 or 2)
    __shared__ __attribute__((aligned(16))) bf16_t As[2][128 * KC];
    __shared__ __attribute__((aligned(16))) bf16_t Bs[2][N * KC];

    const int m0     = blockIdx.x * 128;
    const int tid    = threadIdx.x;
    const int wave   = tid >> 5;
    const int lane   = tid & 31;
    const int laneLo = lane & 15;
    const bool hi    = lane >= 16;
    const int klo    = hi ? 8 : 0;

    v8f acc[NT] = {};

    auto issue_chunk = [&](int kc, int buf) {
#pragma unroll
        for (int j = 0; j < AI; ++j) {
            int e = (j * 256 + tid) * 8;          // flat bf16 index into 128x32 tile
            int row = e >> 5, col = e & 31;
            async_ld_b128((unsigned)(size_t)&As[buf][e],
                          A + (size_t)(m0 + row) * K + kc + col);
        }
#pragma unroll
        for (int j = 0; j < BI; ++j) {
            int e = (j * 256 + tid) * 8;          // flat bf16 index into Nx32 tile
            int n = e >> 5, k2 = e & 31;
            async_ld_b128((unsigned)(size_t)&Bs[buf][e],
                          W + (size_t)n * K + kc + k2);
        }
    };

    issue_chunk(0, 0);
    for (int ki = 0; ki < NCHUNK; ++ki) {
        const int buf = ki & 1;
        if (ki + 1 < NCHUNK) {
            issue_chunk((ki + 1) * KC, buf ^ 1);
            S_WAIT_ASYNCCNT(AI + BI);   // chunk ki retired; chunk ki+1 stays in flight
        } else {
            S_WAIT_ASYNCCNT(0);
        }
        __syncthreads();

        const bf16_t* as = &As[buf][0];
        const bf16_t* bs = &Bs[buf][0];

        FragBF a;
        int mrow = wave * 16 + laneLo;
        a.q[0] = *reinterpret_cast<const uint4*>(as + mrow * KC + klo);
        a.q[1] = *reinterpret_cast<const uint4*>(as + mrow * KC + 16 + klo);

        FragBF bfr[2];
        bfr[0].q[0] = *reinterpret_cast<const uint4*>(bs + (size_t)laneLo * KC + klo);
        bfr[0].q[1] = *reinterpret_cast<const uint4*>(bs + (size_t)laneLo * KC + 16 + klo);
#pragma unroll
        for (int nt = 0; nt < NT; ++nt) {
            if (nt + 1 < NT) {
                int ncol = (nt + 1) * 16 + laneLo;
                bfr[(nt + 1) & 1].q[0] = *reinterpret_cast<const uint4*>(bs + (size_t)ncol * KC + klo);
                bfr[(nt + 1) & 1].q[1] = *reinterpret_cast<const uint4*>(bs + (size_t)ncol * KC + 16 + klo);
            }
            acc[nt] = __builtin_amdgcn_wmma_f32_16x16x32_bf16(
                false, a.v, false, bfr[nt & 1].v, (short)0, acc[nt], false, false);
        }
        __syncthreads();
    }

    // epilogue: C/D layout — VGPR r: lanes0-15 M=r, lanes16-31 M=8+r
    int rowBase = m0 + wave * 16 + (hi ? 8 : 0);
#pragma unroll
    for (int nt = 0; nt < NT; ++nt) {
        int col = nt * 16 + laneLo;
#pragma unroll
        for (int r = 0; r < 8; ++r)
            Y[(size_t)(rowBase + r) * N + col] = acc[nt][r];
    }
}

// ---------------------------------------------------------------------------
// Kernel 5/6: deterministic BN stats (two-stage, no atomics)
// ---------------------------------------------------------------------------
__global__ void bn_partial_kernel(const float* __restrict__ Y, float* __restrict__ partial,
                                  int Mtot, int C) {
    int c = threadIdx.x;                    // blockDim.x == C
    int rowsPer = Mtot / gridDim.x;
    int r0 = blockIdx.x * rowsPer;
    float s = 0.0f, sq = 0.0f;
    for (int r = 0; r < rowsPer; ++r) {
        float v = Y[(size_t)(r0 + r) * C + c];
        s += v; sq += v * v;
    }
    partial[(size_t)blockIdx.x * 2 * C + c]     = s;
    partial[(size_t)blockIdx.x * 2 * C + C + c] = sq;
}

__global__ void bn_finalize_kernel(const float* __restrict__ partial,
                                   const float* __restrict__ g, const float* __restrict__ b,
                                   float* __restrict__ bnp, int nblocks, int C, float invM) {
    int c = threadIdx.x;                    // blockDim.x == C
    float s = 0.0f, sq = 0.0f;
    for (int j = 0; j < nblocks; ++j) {
        s  += partial[(size_t)j * 2 * C + c];
        sq += partial[(size_t)j * 2 * C + C + c];
    }
    float mean  = s * invM;
    float var   = fmaxf(sq * invM - mean * mean, 0.0f);
    float scale = g[c] * rsqrtf(var + BN_EPS);
    bnp[c]     = scale;
    bnp[C + c] = b[c] - mean * scale;
}

// ---------------------------------------------------------------------------
// Kernel 7: BN + ReLU -> bf16 (input to GEMM2)     Kernel 8: BN + ReLU -> f32
// ---------------------------------------------------------------------------
__global__ void bn_relu_bf16_kernel(const float* __restrict__ Y, const float* __restrict__ bnp,
                                    bf16_t* __restrict__ O, int total, int Cmask, int C) {
    int i = blockIdx.x * blockDim.x + threadIdx.x;
    if (i >= total) return;
    int c = i & Cmask;
    float v = fmaf(Y[i], bnp[c], bnp[C + c]);
    O[i] = f2bf(fmaxf(v, 0.0f));
}

__global__ void bn_relu_f32_kernel(const float* __restrict__ Y, const float* __restrict__ bnp,
                                   float* __restrict__ O, int total, int Cmask, int C) {
    int i = blockIdx.x * blockDim.x + threadIdx.x;
    if (i >= total) return;
    int c = i & Cmask;
    float v = fmaf(Y[i], bnp[c], bnp[C + c]);
    O[i] = fmaxf(v, 0.0f);
}

// ---------------------------------------------------------------------------
extern "C" void kernel_launch(void* const* d_in, const int* in_sizes, int n_in,
                              void* d_out, int out_size, void* d_ws, size_t ws_size,
                              hipStream_t stream) {
    const int B = 8, N = 8192, S = 2048, D1 = 128, D2 = 256, Cin = 384;
    const int M = B * N;                       // 65536 points
    (void)in_sizes; (void)n_in; (void)out_size; (void)ws_size; (void)S;

    const float* xyz1    = (const float*)d_in[0];
    const float* xyz2    = (const float*)d_in[1];
    const float* points1 = (const float*)d_in[2];
    const float* points2 = (const float*)d_in[3];
    const float* W1      = (const float*)d_in[4];
    const float* g1      = (const float*)d_in[5];
    const float* b1      = (const float*)d_in[6];
    const float* W2      = (const float*)d_in[7];
    const float* g2      = (const float*)d_in[8];
    const float* b2      = (const float*)d_in[9];
    float* out = (float*)d_out;

    // --- workspace layout (256B aligned) ---
    char* ws = (char*)d_ws;
    size_t o = 0;
    auto alloc = [&](size_t bytes) { size_t r = o; o += (bytes + 255) & ~(size_t)255; return r; };
    size_t oXbf   = alloc((size_t)M * Cin * 2);   // 50.3 MB  bf16 X (also reused for Y2raw)
    size_t oY1    = alloc((size_t)M * 256 * 4);   // 67.1 MB  f32 Y1 raw
    size_t oY1bf  = alloc((size_t)M * 256 * 2);   // 33.6 MB  bf16 Y1 post-BN
    size_t oW1b   = alloc((size_t)256 * 384 * 2);
    size_t oW2b   = alloc((size_t)128 * 256 * 2);
    size_t oIdx   = alloc((size_t)M * 3 * 4);
    size_t oWgt   = alloc((size_t)M * 3 * 4);
    size_t oPart  = alloc((size_t)256 * 2 * 256 * 4);
    size_t oBnp1  = alloc((size_t)2 * 256 * 4);
    size_t oBnp2  = alloc((size_t)2 * 128 * 4);

    bf16_t* Xbf   = (bf16_t*)(ws + oXbf);
    float*  Y1    = (float*)(ws + oY1);
    bf16_t* Y1bf  = (bf16_t*)(ws + oY1bf);
    float*  Y2    = (float*)(ws + oXbf);          // reuse X region (dead after GEMM1)
    bf16_t* W1b   = (bf16_t*)(ws + oW1b);
    bf16_t* W2b   = (bf16_t*)(ws + oW2b);
    int*    idx   = (int*)(ws + oIdx);
    float*  wgt   = (float*)(ws + oWgt);
    float*  part  = (float*)(ws + oPart);
    float*  bnp1  = (float*)(ws + oBnp1);
    float*  bnp2  = (float*)(ws + oBnp2);

    // 1) weights -> bf16
    cvt_weights_kernel<<<(256 * 384 + 255) / 256, 256, 0, stream>>>(W1, W2, W1b, W2b);
    // 2) 3-NN search
    knn3_kernel<<<B * (N / 256), 256, 0, stream>>>(xyz1, xyz2, idx, wgt);
    // 3) interpolate + concat -> bf16 X[M,384]
    interp_concat_kernel<<<M, 256, 0, stream>>>(points1, points2, idx, wgt, Xbf);
    // 4) GEMM1: Y1[M,256] = X[M,384] * W1[256,384]^T
    gemm_bf16_wmma_kernel<384, 16><<<M / 128, 256, 0, stream>>>(Xbf, W1b, Y1);
    // 5) BN1 stats + params
    bn_partial_kernel<<<256, 256, 0, stream>>>(Y1, part, M, 256);
    bn_finalize_kernel<<<1, 256, 0, stream>>>(part, g1, b1, bnp1, 256, 256, 1.0f / (float)M);
    // 6) BN1 + ReLU -> bf16
    bn_relu_bf16_kernel<<<(M * 256) / 256, 256, 0, stream>>>(Y1, bnp1, Y1bf, M * 256, 255, 256);
    // 7) GEMM2: Y2[M,128] = Y1bf[M,256] * W2[128,256]^T
    gemm_bf16_wmma_kernel<256, 8><<<M / 128, 256, 0, stream>>>(Y1bf, W2b, Y2);
    // 8) BN2 stats + params
    bn_partial_kernel<<<256, 128, 0, stream>>>(Y2, part, M, 128);
    bn_finalize_kernel<<<1, 128, 0, stream>>>(part, g2, b2, bnp2, 256, 128, 1.0f / (float)M);
    // 9) BN2 + ReLU -> f32 output
    bn_relu_f32_kernel<<<(M * 128) / 256, 256, 0, stream>>>(Y2, bnp2, out, M * 128, 127, 128);
    (void)D1; (void)D2;
}
